// ConcatenateAttention_80281528697086
// MI455X (gfx1250) — matrix operand order repaired and checked by reading, not compiled, not run on hardware
//
#include <hip/hip_runtime.h>
#include <hip/hip_bf16.h>

typedef __attribute__((ext_vector_type(16))) __bf16 v16bf;
typedef __attribute__((ext_vector_type(8)))  __bf16 v8bf;
typedef __attribute__((ext_vector_type(8)))  float  v8f;

#define NB 32
#define ND 512
#define NT 4096
#define NH 512
#define TT 32            // t-columns per block
#define KSTRIDE 520      // padded bf16 k-stride in LDS ([t][k] layout)

// ---------------------------------------------------------------------------
// Prep 1: Wk bf16 = w42[:, :D]   (512x512 bf16, L2-resident afterwards)
// ---------------------------------------------------------------------------
__global__ __launch_bounds__(256) void ca_prep_wk(const float* __restrict__ w42,
                                                  __bf16* __restrict__ Wk) {
  int idx = blockIdx.x * 256 + threadIdx.x;       // 0 .. 262143
  int h = idx >> 9, k = idx & 511;
  Wk[idx] = (__bf16)w42[(size_t)h * 1024 + k];
}

// ---------------------------------------------------------------------------
// Prep 2: qb[b,h] = sum_d w42[h, D+d] * query[b,d] + b4[h]
// ---------------------------------------------------------------------------
__global__ __launch_bounds__(256) void ca_prep_qb(const float* __restrict__ w42,
                                                  const float* __restrict__ query,
                                                  const float* __restrict__ b4,
                                                  float* __restrict__ qb) {
  int idx = blockIdx.x * 256 + threadIdx.x;       // 0 .. 16383 ; b-major
  int b = idx >> 9, h = idx & 511;
  const float4* wr = (const float4*)(w42 + (size_t)h * 1024 + 512);
  const float4* qv = (const float4*)(query + (size_t)b * 512);
  float s = b4[h];
#pragma unroll 4
  for (int i = 0; i < 128; ++i) {
    float4 a = wr[i], c = qv[i];
    s += a.x * c.x + a.y * c.y + a.z * c.z + a.w * c.w;
  }
  qb[idx] = s;
}

// ---------------------------------------------------------------------------
// Main: per (batch, 32-col tile) compute
//   score[t] = sum_h w54[h] * tanh( (Wk @ keys[b])[h,t] + qb[b,h] )
// via v_wmma_f32_16x16x32_bf16, keys tile staged once in LDS as bf16 [t][k].
// 8 waves = 4 h-groups (128 h each) x 2 t-subtiles (16 cols each).
// ---------------------------------------------------------------------------
__global__ __launch_bounds__(256) void ca_scores(const float* __restrict__ keys,
                                                 const __bf16* __restrict__ Wk,
                                                 const float* __restrict__ qb,
                                                 const float* __restrict__ w54,
                                                 float* __restrict__ sc) {
  __shared__ __bf16 ldsK[TT * KSTRIDE];   // ~33 KB
  __shared__ float  ldsS[TT];

  const int b    = blockIdx.y;
  const int t0   = blockIdx.x * TT;
  const int tid  = threadIdx.x;
  const int lane = tid & 31;

  // ---- stage keys[b, :, t0..t0+31] -> bf16 LDS, [t][k] (transposed) ----
  {
    const int t  = tid & 31;          // coalesced along t within a wave
    const int d0 = tid >> 5;          // 0..7
    const float* kp = keys + (size_t)b * ND * NT + t0 + t;
#pragma unroll 4
    for (int i = 0; i < ND / 8; ++i) {
      int d = d0 + i * 8;
      ldsK[t * KSTRIDE + d] = (__bf16)kp[(size_t)d * NT];
    }
  }
  if (tid < TT) ldsS[tid] = 0.0f;
  __syncthreads();

  const int wave = tid >> 5;
  const int g    = wave >> 1;         // h-group 0..3
  const int tt   = wave & 1;          // t-subtile 0..1
  const int hi   = lane >> 4;         // lane half (WMMA layout)
  const int ln   = lane & 15;

  // B-matrix base: lane ln = column N, lane-half selects K+8 (32x16 bf16 B)
  const __bf16* ldsB = ldsK + (tt * 16 + ln) * KSTRIDE + hi * 8;

  float s = 0.0f;
  for (int ht = 0; ht < 8; ++ht) {
    const int h0 = g * 128 + ht * 16;
    // A-matrix: lane ln = row M; 16-bit A 16x32 layout: lane-half owns K hi*8..
    const __bf16* Arow = Wk + (size_t)(h0 + ln) * ND + hi * 8;
    v8f acc = {};
#pragma unroll
    for (int kc = 0; kc < 16; ++kc) {
      v8bf alo = *(const v8bf*)(Arow + kc * 32);        // K = kc*32+hi*8+0..7
      v8bf ahi = *(const v8bf*)(Arow + kc * 32 + 16);   // K = kc*32+16+hi*8+0..7
      v16bf A = __builtin_shufflevector(alo, ahi,
          0, 1, 2, 3, 4, 5, 6, 7, 8, 9, 10, 11, 12, 13, 14, 15);
      v8bf blo = *(const v8bf*)(ldsB + kc * 32);        // K = kc*32+hi*8+0..7
      v8bf bhi = *(const v8bf*)(ldsB + kc * 32 + 16);   // K = kc*32+16+hi*8+0..7
      v16bf Bv = __builtin_shufflevector(blo, bhi,
          0, 1, 2, 3, 4, 5, 6, 7, 8, 9, 10, 11, 12, 13, 14, 15);
      acc = __builtin_amdgcn_wmma_f32_16x16x32_bf16(
          false, A, false, Bv, (short)0, acc, false, false);
    }
    // C/D layout: element r -> M = r + 8*hi, N = ln. h rows are consecutive.
    const float4* qv4 = (const float4*)(qb + (size_t)b * NH + h0 + hi * 8);
    const float4* wv4 = (const float4*)(w54 + h0 + hi * 8);
    float4 q0 = qv4[0], q1 = qv4[1];
    float4 w0 = wv4[0], w1 = wv4[1];
    s += w0.x * tanhf(acc[0] + q0.x);
    s += w0.y * tanhf(acc[1] + q0.y);
    s += w0.z * tanhf(acc[2] + q0.z);
    s += w0.w * tanhf(acc[3] + q0.w);
    s += w1.x * tanhf(acc[4] + q1.x);
    s += w1.y * tanhf(acc[5] + q1.y);
    s += w1.z * tanhf(acc[6] + q1.z);
    s += w1.w * tanhf(acc[7] + q1.w);
  }
  // Reduce over h-groups and lane halves: ds_add_f32 into per-column score.
  atomicAdd(&ldsS[tt * 16 + ln], s);
  __syncthreads();
  if (tid < TT) sc[(size_t)b * NT + t0 + tid] = ldsS[tid];
}

// ---------------------------------------------------------------------------
// Softmax over T per batch, in place (b5 is a constant shift -> cancels).
// ---------------------------------------------------------------------------
__global__ __launch_bounds__(256) void ca_softmax(float* __restrict__ sc) {
  __shared__ float red[256];
  const int b = blockIdx.x, tid = threadIdx.x;
  float* s = sc + (size_t)b * NT;
  float m = -1e30f;
  for (int i = tid; i < NT; i += 256) m = fmaxf(m, s[i]);
  red[tid] = m;
  __syncthreads();
  for (int o = 128; o > 0; o >>= 1) {
    if (tid < o) red[tid] = fmaxf(red[tid], red[tid + o]);
    __syncthreads();
  }
  m = red[0];
  __syncthreads();
  float sum = 0.0f;
  for (int i = tid; i < NT; i += 256) sum += __expf(s[i] - m);
  red[tid] = sum;
  __syncthreads();
  for (int o = 128; o > 0; o >>= 1) {
    if (tid < o) red[tid] += red[tid + o];
    __syncthreads();
  }
  float inv = 1.0f / red[0];
  for (int i = tid; i < NT; i += 256) s[i] = __expf(s[i] - m) * inv;
}

// ---------------------------------------------------------------------------
// Output: a6[b,d] = sum_t values[b,d,t] * p[b,t]  (streaming GEMV, float4)
// One wave per d-row; 8 rows per block.
// ---------------------------------------------------------------------------
__global__ __launch_bounds__(256) void ca_out(const float* __restrict__ values,
                                              const float* __restrict__ probs,
                                              float* __restrict__ out) {
  const int b    = blockIdx.y;
  const int lane = threadIdx.x & 31;
  const int d    = blockIdx.x * 8 + (threadIdx.x >> 5);
  const float4* vp = (const float4*)(values + ((size_t)b * ND + d) * NT);
  const float4* pp = (const float4*)(probs + (size_t)b * NT);
  float acc = 0.0f;
#pragma unroll 4
  for (int i = 0; i < 32; ++i) {
    float4 v = vp[i * 32 + lane];
    float4 p = pp[i * 32 + lane];
    acc += v.x * p.x + v.y * p.y + v.z * p.z + v.w * p.w;
  }
#pragma unroll
  for (int o = 16; o > 0; o >>= 1) acc += __shfl_xor(acc, o, 32);
  if (lane == 0) out[(size_t)b * ND + d] = acc;
}

// ---------------------------------------------------------------------------
extern "C" void kernel_launch(void* const* d_in, const int* in_sizes, int n_in,
                              void* d_out, int out_size, void* d_ws, size_t ws_size,
                              hipStream_t stream) {
  const float* query  = (const float*)d_in[0];
  const float* keys   = (const float*)d_in[1];
  const float* values = (const float*)d_in[2];
  const float* w42    = (const float*)d_in[3];
  const float* b4     = (const float*)d_in[4];
  const float* w54    = (const float*)d_in[5];
  // d_in[6] = b5 : constant shift before softmax, cancels -> unused.
  float* out = (float*)d_out;

  char* ws = (char*)d_ws;
  __bf16* Wk = (__bf16*)ws;                               // 512 KB
  float*  qb = (float*)(ws + 512 * 1024);                 // 64 KB
  float*  sc = (float*)(ws + 512 * 1024 + 64 * 1024);     // 512 KB (scores->probs)

  ca_prep_wk<<<dim3((NH * ND) / 256), dim3(256), 0, stream>>>(w42, Wk);
  ca_prep_qb<<<dim3((NB * NH) / 256), dim3(256), 0, stream>>>(w42, query, b4, qb);
  ca_scores<<<dim3(NT / TT, NB), dim3(256), 0, stream>>>(keys, Wk, qb, w54, sc);
  ca_softmax<<<dim3(NB), dim3(256), 0, stream>>>(sc);
  ca_out<<<dim3(ND / 8, NB), dim3(256), 0, stream>>>(values, sc, out);
}
